// EpisodicMemory_84293028151863
// MI455X (gfx1250) — matrix operand order repaired and checked by reading, not compiled, not run on hardware
//
#include <hip/hip_runtime.h>

#define KD   256
#define VD   257
#define NS   64
#define VNP  272   // 17*16  padded N for value-dim GEMMs
#define KVP  288   // 9*32   padded K for retrieved@Wv

typedef __attribute__((ext_vector_type(16))) __bf16 v16bf;
typedef __attribute__((ext_vector_type(8)))  __bf16 v8bf;
typedef __attribute__((ext_vector_type(8)))  float  v8f;
typedef __attribute__((ext_vector_type(4)))  float  v4f;

#define WMMA_BF16(acc, a, b) \
  __builtin_amdgcn_wmma_f32_16x16x32_bf16(false, (a), false, (b), (short)0, (acc), false, false)

// ---------------- dynamic LDS layout (bytes) --------------------------------
#define OFF_WBUF0   0            // weight staging double buffer (9216 each)
#define OFF_WBUF1   9216
#define OFF_GLN     18432        // 256 f32
#define OFF_BLN     19456        // 256 f32
#define OFF_WAVE    20480        // 8 per-wave regions
#define WOFF_HB     0            // raw h bf16 [16][256] (8192B); attention overlay [16][64]
#define WOFF_QB     8192         // query bf16 [16][256] (8192B)
#define WOFF_RB     16384        // retrieved bf16 [16][288] (9216B)
#define WOFF_STATS  25600        // mu[16] rstd[16] qinv[16] gate[16] f32 (256B)
#define WAVE_BYTES  25856
#define SMEM_TOTAL  (OFF_WAVE + 8 * WAVE_BYTES)   // 227328 B

// ---------------- CDNA5 async LDS staging -----------------------------------
static __device__ __forceinline__ void async_b128(unsigned lds_off, const void* g) {
  asm volatile("global_load_async_to_lds_b128 %0, %1, off"
               :: "v"(lds_off), "v"((unsigned long long)(uintptr_t)g) : "memory");
}
static __device__ __forceinline__ void async_b64(unsigned lds_off, const void* g) {
  asm volatile("global_load_async_to_lds_b64 %0, %1, off"
               :: "v"(lds_off), "v"((unsigned long long)(uintptr_t)g) : "memory");
}
static __device__ __forceinline__ void wait_async0() { asm volatile("s_wait_asynccnt 0x0" ::: "memory"); }
static __device__ __forceinline__ void wait_async1() { asm volatile("s_wait_asynccnt 0x1" ::: "memory"); }
static __device__ __forceinline__ void wait_async2() { asm volatile("s_wait_asynccnt 0x2" ::: "memory"); }

// ---------------- WMMA operand builders (wave32 layouts) --------------------
// B fragment: lane holds col N = (lane&15); K = kbase + (e>>3)*16 + half*8 + (e&7)
static __device__ __forceinline__ v16bf load_b_frag(const __bf16* Wt, int n, int stride,
                                                    int kbase, int half) {
  const __bf16* p0 = Wt + (size_t)n * stride + kbase + half * 8;
  v8bf lo = *(const v8bf*)p0;
  v8bf hi = *(const v8bf*)(p0 + 16);
  v16bf b;
#pragma unroll
  for (int i = 0; i < 8; ++i) { b[i] = lo[i]; b[i + 8] = hi[i]; }
  return b;
}
// A fragment: lane holds row M = lane&15; K = kbase + (e>>3)*16 + half*8 + (e&7)
static __device__ __forceinline__ v16bf load_a_bf16(const __bf16* base, int row, int stride,
                                                    int kbase, int half) {
  const __bf16* p0 = base + (size_t)row * stride + kbase + half * 8;
  v8bf lo = *(const v8bf*)p0;
  v8bf hi = *(const v8bf*)(p0 + 16);
  v16bf a;
#pragma unroll
  for (int i = 0; i < 8; ++i) { a[i] = lo[i]; a[i + 8] = hi[i]; }
  return a;
}
static __device__ __forceinline__ v16bf load_a_f32(const float* base, int row, int stride,
                                                   int kbase, int half) {
  const float* p0 = base + (size_t)row * stride + kbase + half * 8;
  v4f x0 = *(const v4f*)p0,        x1 = *(const v4f*)(p0 + 4);
  v4f y0 = *(const v4f*)(p0 + 16), y1 = *(const v4f*)(p0 + 20);
  v16bf a;
#pragma unroll
  for (int i = 0; i < 4; ++i) {
    a[i]      = (__bf16)x0[i];  a[i + 4]  = (__bf16)x1[i];
    a[i + 8]  = (__bf16)y0[i];  a[i + 12] = (__bf16)y1[i];
  }
  return a;
}

// ---------------- weight prep -----------------------------------------------
__global__ void ep_transpose256(const float* __restrict__ W, __bf16* __restrict__ Wt) {
  int n = blockIdx.x, k = threadIdx.x;
  Wt[(size_t)n * KD + k] = (__bf16)W[(size_t)k * KD + n];
}
__global__ void ep_prep_wvt(const float* __restrict__ Wv, __bf16* __restrict__ Wvt) {
  int n = blockIdx.x;
  for (int k = threadIdx.x; k < KVP; k += blockDim.x) {
    float v = (n < VD && k < VD) ? Wv[(size_t)k * VD + n] : 0.f;
    Wvt[(size_t)n * KVP + k] = (__bf16)v;
  }
}
__global__ void ep_prep_valt(const float* __restrict__ values, __bf16* __restrict__ valt) {
  int n = blockIdx.x, s = threadIdx.x;
  valt[(size_t)n * NS + s] = (__bf16)((n < VD) ? values[(size_t)s * VD + n] : 0.f);
}
__global__ void ep_prep_kn(const float* __restrict__ keys, __bf16* __restrict__ knb) {
  __shared__ float red[64];
  int s = blockIdx.x, t = threadIdx.x;
  float ss = 0.f;
  for (int i = t; i < KD; i += 64) { float v = keys[(size_t)s * KD + i]; ss += v * v; }
  red[t] = ss; __syncthreads();
  for (int st = 32; st > 0; st >>= 1) { if (t < st) red[t] += red[t + st]; __syncthreads(); }
  float inv = rsqrtf(red[0] + 1e-12f);
  for (int i = t; i < KD; i += 64) knb[(size_t)s * KD + i] = (__bf16)(keys[(size_t)s * KD + i] * inv);
}

// ---------------- fused pipeline: 8 waves x 16 rows, LDS-staged weights -----
__global__ __launch_bounds__(256) void EpisodicMemory_84293028151863_kernel(
    const float* __restrict__ x,  const float* __restrict__ b1,
    const float* __restrict__ gln, const float* __restrict__ bln,
    const float* __restrict__ b2, const float* __restrict__ bv,
    const float* __restrict__ Wg, const float* __restrict__ bg,
    const __bf16* __restrict__ W1t, const __bf16* __restrict__ W2t,
    const __bf16* __restrict__ Wvt, const __bf16* __restrict__ knb,
    const __bf16* __restrict__ valt,
    float* __restrict__ out0, float* __restrict__ out1,
    int* __restrict__ out2, float* __restrict__ out3) {
  extern __shared__ char smem[];
  const int tid  = threadIdx.x;
  const int lane = tid & 31, wv = tid >> 5;
  const int half = lane >> 4, lm = lane & 15;
  const int rowbase = blockIdx.x * 128 + wv * 16;
  const unsigned dynbase = __builtin_amdgcn_groupstaticsize();

  char* wavebase = smem + OFF_WAVE + wv * WAVE_BYTES;
  __bf16* sHb  = (__bf16*)(wavebase + WOFF_HB);     // raw h / attention overlay
  __bf16* sQb  = (__bf16*)(wavebase + WOFF_QB);
  __bf16* sRb  = (__bf16*)(wavebase + WOFF_RB);
  float*  sMu   = (float*)(wavebase + WOFF_STATS);
  float*  sRstd = sMu + 16;
  float*  sQinv = sMu + 32;
  float*  sGate = sMu + 48;
  float*  sGln  = (float*)(smem + OFF_GLN);
  float*  sBln  = (float*)(smem + OFF_BLN);

  // double-buffer pointer (computed, never aggregate-initialized: keeps the
  // LDS addrspacecast out of static initializers)
  auto wptr = [&](int buf) -> const __bf16* {
    return (const __bf16*)(smem + (buf ? OFF_WBUF1 : OFF_WBUF0));
  };

  // cooperative stage of LN affine params
  sGln[tid] = gln[tid];
  sBln[tid] = bln[tid];

  // async-stage one contiguous 8KB chunk (16 rows x 256 bf16)
  auto stage8k = [&](const __bf16* W, int chunk, int buf) {
    unsigned dst = dynbase + (buf ? OFF_WBUF1 : OFF_WBUF0) + tid * 16;
    const char* src = (const char*)W + (size_t)chunk * 8192 + tid * 16;
    async_b128(dst, src);
    async_b128(dst + 4096, src + 4096);
  };
  auto stage_valt = [&](int chunk, int buf) {          // 2KB chunk
    unsigned dst = dynbase + (buf ? OFF_WBUF1 : OFF_WBUF0) + tid * 8;
    const char* src = (const char*)valt + (size_t)chunk * 2048 + tid * 8;
    async_b64(dst, src);
  };
  auto stage_wvt = [&](int chunk, int buf) {           // 9216B chunk
    unsigned dbase = dynbase + (buf ? OFF_WBUF1 : OFF_WBUF0);
    const char* src = (const char*)Wvt + (size_t)chunk * 9216;
    for (int i = tid; i < 576; i += 256) async_b128(dbase + i * 16, src + (size_t)i * 16);
  };

  v16bf afr[8];

  // ===== Stage 1: h = x @ W1 + b1  (raw h -> sHb bf16; LN stats in regs) ====
#pragma unroll
  for (int kc = 0; kc < 8; ++kc)
    afr[kc] = load_a_f32(x + (size_t)rowbase * KD, lm, KD, kc * 32, half);
  float hsum[8] = {}, hss[8] = {};
  stage8k(W1t, 0, 0);
  for (int nt = 0; nt < 16; ++nt) {
    if (nt + 1 < 16) { stage8k(W1t, nt + 1, (nt + 1) & 1); wait_async2(); }
    else wait_async0();
    __syncthreads();
    const __bf16* bw = wptr(nt & 1);
    v8f acc = {};
#pragma unroll
    for (int kc = 0; kc < 8; ++kc)
      acc = WMMA_BF16(acc, afr[kc], load_b_frag(bw, lm, KD, kc * 32, half));
    float bias = b1[nt * 16 + lm];
#pragma unroll
    for (int r = 0; r < 8; ++r) {
      float h = acc[r] + bias;
      sHb[(r + half * 8) * KD + nt * 16 + lm] = (__bf16)h;
      hsum[r] += h; hss[r] += h * h;
    }
    __syncthreads();
  }
#pragma unroll
  for (int r = 0; r < 8; ++r) {
#pragma unroll
    for (int mk = 1; mk < 16; mk <<= 1) {
      hsum[r] += __shfl_xor(hsum[r], mk, 32);
      hss[r]  += __shfl_xor(hss[r],  mk, 32);
    }
  }
  if (lm == 0) {
#pragma unroll
    for (int r = 0; r < 8; ++r) {
      float mu  = hsum[r] * (1.f / 256.f);
      float var = hss[r] * (1.f / 256.f) - mu * mu;
      sMu[r + half * 8]   = mu;
      sRstd[r + half * 8] = rsqrtf(var + 1e-5f);
    }
  }
  __syncthreads();

  // ===== Stage 2+3: A = GELU(LN(h)); query = A @ W2 + b2 ====================
  {
    float mu = sMu[lm], rs = sRstd[lm];
#pragma unroll
    for (int kc = 0; kc < 8; ++kc) {
      v16bf a;
#pragma unroll
      for (int e = 0; e < 16; ++e) {
        int k = kc * 32 + ((e >> 3) << 4) + half * 8 + (e & 7);
        float v = (float)sHb[lm * KD + k];
        v = (v - mu) * rs * sGln[k] + sBln[k];
        a[e] = (__bf16)(0.5f * v * (1.f + erff(v * 0.70710678f)));
      }
      afr[kc] = a;
    }
  }
  float qss[8] = {};
  stage8k(W2t, 0, 0);
  for (int nt = 0; nt < 16; ++nt) {
    if (nt + 1 < 16) { stage8k(W2t, nt + 1, (nt + 1) & 1); wait_async2(); }
    else wait_async0();
    __syncthreads();
    const __bf16* bw = wptr(nt & 1);
    v8f acc = {};
#pragma unroll
    for (int kc = 0; kc < 8; ++kc)
      acc = WMMA_BF16(acc, afr[kc], load_b_frag(bw, lm, KD, kc * 32, half));
    float bias = b2[nt * 16 + lm];
#pragma unroll
    for (int r = 0; r < 8; ++r) {
      float q = acc[r] + bias;
      sQb[(r + half * 8) * KD + nt * 16 + lm] = (__bf16)q;
      qss[r] += q * q;
    }
    __syncthreads();
  }
#pragma unroll
  for (int r = 0; r < 8; ++r) {
#pragma unroll
    for (int mk = 1; mk < 16; mk <<= 1) qss[r] += __shfl_xor(qss[r], mk, 32);
  }
  if (lm == 0) {
#pragma unroll
    for (int r = 0; r < 8; ++r) sQinv[r + half * 8] = rsqrtf(qss[r] + 1e-12f);
  }
  __syncthreads();

  // ===== Stage 5: similarity, softmax, argmax, sim_max ======================
#pragma unroll
  for (int kc = 0; kc < 8; ++kc) afr[kc] = load_a_bf16(sQb, lm, KD, kc * 32, half);
  v8f st[4];
  stage8k(knb, 0, 0);
  for (int t = 0; t < 4; ++t) {
    if (t + 1 < 4) { stage8k(knb, t + 1, (t + 1) & 1); wait_async2(); }
    else wait_async0();
    __syncthreads();
    const __bf16* bw = wptr(t & 1);
    v8f acc = {};
#pragma unroll
    for (int kc = 0; kc < 8; ++kc)
      acc = WMMA_BF16(acc, afr[kc], load_b_frag(bw, lm, KD, kc * 32, half));
    st[t] = acc;
    __syncthreads();
  }
  __bf16* sAtt = sHb;   // overlay: [16][64] attention
#pragma unroll
  for (int r = 0; r < 8; ++r) {
    int row = r + half * 8;
    float qir = sQinv[row];
    float v0 = st[0][r] * qir, v1 = st[1][r] * qir, v2 = st[2][r] * qir, v3 = st[3][r] * qir;
    float m = v0; int idx = lm;
    if (v1 > m) { m = v1; idx = 16 + lm; }
    if (v2 > m) { m = v2; idx = 32 + lm; }
    if (v3 > m) { m = v3; idx = 48 + lm; }
#pragma unroll
    for (int mk = 1; mk < 16; mk <<= 1) {
      float om = __shfl_xor(m, mk, 32); int oi = __shfl_xor(idx, mk, 32);
      if (om > m || (om == m && oi < idx)) { m = om; idx = oi; }
    }
    float e0 = __expf(v0 - m), e1 = __expf(v1 - m), e2 = __expf(v2 - m), e3 = __expf(v3 - m);
    float sum = e0 + e1 + e2 + e3;
#pragma unroll
    for (int mk = 1; mk < 16; mk <<= 1) sum += __shfl_xor(sum, mk, 32);
    float inv = 1.f / sum;
    sAtt[row * NS + lm]      = (__bf16)(e0 * inv);
    sAtt[row * NS + 16 + lm] = (__bf16)(e1 * inv);
    sAtt[row * NS + 32 + lm] = (__bf16)(e2 * inv);
    sAtt[row * NS + 48 + lm] = (__bf16)(e3 * inv);
    if (lm == 0) { out3[rowbase + row] = m; out2[rowbase + row] = idx; }
  }

  // ===== Stage 6: retrieved = attention @ values -> sRb (padded) ============
  stage_valt(0, 0);
  v16bf aat0 = load_a_bf16(sAtt, lm, NS, 0, half);
  v16bf aat1 = load_a_bf16(sAtt, lm, NS, 32, half);
  for (int nt = 0; nt < 17; ++nt) {
    if (nt + 1 < 17) { stage_valt(nt + 1, (nt + 1) & 1); wait_async1(); }
    else wait_async0();
    __syncthreads();
    const __bf16* bw = wptr(nt & 1);
    v8f acc = {};
    acc = WMMA_BF16(acc, aat0, load_b_frag(bw, lm, NS, 0, half));
    acc = WMMA_BF16(acc, aat1, load_b_frag(bw, lm, NS, 32, half));
#pragma unroll
    for (int r = 0; r < 8; ++r) sRb[(r + half * 8) * KVP + nt * 16 + lm] = (__bf16)acc[r];
    __syncthreads();
  }
#pragma unroll
  for (int r = 0; r < 8; ++r) sRb[(r + half * 8) * KVP + 272 + lm] = (__bf16)0.f;

  // ===== Gate: sigmoid([query, retrieved] @ Wg + bg) ========================
  {
    float partial = 0.f;
    if (half == 0) {
      for (int k = 0; k < KD; ++k) partial += (float)sQb[lm * KD + k] * Wg[k];
    } else {
      for (int n = 0; n < VD; ++n) partial += (float)sRb[lm * KVP + n] * Wg[KD + n];
    }
    partial += __shfl_xor(partial, 16, 32);
    float g = 1.f / (1.f + __expf(-(partial + bg[0])));
    if (half == 0) { sGate[lm] = g; out1[rowbase + lm] = g; }
  }
  __syncthreads();

  // ===== Stage 7: out0 = tanh(retrieved @ Wv + bv) * gate ===================
  v16bf afr9[9];
#pragma unroll
  for (int kc = 0; kc < 9; ++kc) afr9[kc] = load_a_bf16(sRb, lm, KVP, kc * 32, half);
  float gr[8];
#pragma unroll
  for (int r = 0; r < 8; ++r) gr[r] = sGate[r + half * 8];
  stage_wvt(0, 0);
  for (int nt = 0; nt < 17; ++nt) {
    if (nt + 1 < 17) { stage_wvt(nt + 1, (nt + 1) & 1); wait_async2(); }
    else wait_async0();
    __syncthreads();
    const __bf16* bw = wptr(nt & 1);
    v8f acc = {};
#pragma unroll
    for (int kc = 0; kc < 9; ++kc)
      acc = WMMA_BF16(acc, afr9[kc], load_b_frag(bw, lm, KVP, kc * 32, half));
    int n = nt * 16 + lm;
    if (n < VD) {
      float bias = bv[n];
#pragma unroll
      for (int r = 0; r < 8; ++r) {
        int row = rowbase + r + half * 8;
        out0[(size_t)row * VD + n] = tanhf(acc[r] + bias) * gr[r];
      }
    }
    __syncthreads();
  }
}

extern "C" void kernel_launch(void* const* d_in, const int* in_sizes, int n_in,
                              void* d_out, int out_size, void* d_ws, size_t ws_size,
                              hipStream_t stream) {
  const float* x      = (const float*)d_in[0];
  const float* keys   = (const float*)d_in[1];
  const float* values = (const float*)d_in[2];
  const float* W1     = (const float*)d_in[3];
  const float* b1     = (const float*)d_in[4];
  const float* gln    = (const float*)d_in[5];
  const float* bln    = (const float*)d_in[6];
  const float* W2     = (const float*)d_in[7];
  const float* b2     = (const float*)d_in[8];
  const float* Wv     = (const float*)d_in[9];
  const float* bv     = (const float*)d_in[10];
  const float* Wg     = (const float*)d_in[11];
  const float* bg     = (const float*)d_in[12];

  const int B = in_sizes[0] / KD;   // 131072

  char* ws = (char*)d_ws;
  __bf16* W1t  = (__bf16*)(ws);
  __bf16* W2t  = (__bf16*)(ws + 131072);
  __bf16* Wvt  = (__bf16*)(ws + 262144);
  __bf16* knb  = (__bf16*)(ws + 262144 + 156672);
  __bf16* valt = (__bf16*)(ws + 262144 + 156672 + 32768);

  float* out0 = (float*)d_out;
  float* out1 = out0 + (size_t)B * VD;
  int*   out2 = (int*)(out0 + (size_t)B * (VD + 1));
  float* out3 = out0 + (size_t)B * (VD + 2);

  ep_transpose256<<<KD, KD, 0, stream>>>(W1, W1t);
  ep_transpose256<<<KD, KD, 0, stream>>>(W2, W2t);
  ep_prep_wvt<<<VNP, 256, 0, stream>>>(Wv, Wvt);
  ep_prep_kn<<<NS, 64, 0, stream>>>(keys, knb);
  ep_prep_valt<<<VNP, NS, 0, stream>>>(values, valt);

  EpisodicMemory_84293028151863_kernel<<<B / 128, 256, SMEM_TOTAL, stream>>>(
      x, b1, gln, bln, b2, bv, Wg, bg, W1t, W2t, Wvt, knb, valt,
      out0, out1, out2, out3);
}
